// WeightedEnsembleAggregator_69509750718524
// MI455X (gfx1250) — matrix operand order repaired and checked by reading, not compiled, not run on hardware
//
#include <hip/hip_runtime.h>

typedef __attribute__((ext_vector_type(16))) _Float16 v16h;
typedef __attribute__((ext_vector_type(8)))  float    v8f;
typedef __attribute__((ext_vector_type(4)))  float    vf4;   // clang-native float4

#define DIM 128   // feature dim D == hidden dim H == 128

// ---------------------------------------------------------------------------
// Kernel 0: zero the output accumulator and the denominator scratch.
// ---------------------------------------------------------------------------
__global__ void zero_kernel(float* __restrict__ out, long n_out,
                            float* __restrict__ denom, int n_den) {
  long i = (long)blockIdx.x * blockDim.x + threadIdx.x;
  if (i < n_out) out[i] = 0.0f;
  if (i < n_den) denom[i] = 0.0f;
}

// Load 8 consecutive f32 (non-temporal: pure stream) and convert into
// halves [base, base+8) of a v16h.
__device__ __forceinline__ void load8_cvt_nt(v16h& a, int base,
                                             const float* __restrict__ p) {
  vf4 x0 = __builtin_nontemporal_load(reinterpret_cast<const vf4*>(p));
  vf4 x1 = __builtin_nontemporal_load(reinterpret_cast<const vf4*>(p + 4));
  a[base + 0] = (_Float16)x0[0]; a[base + 1] = (_Float16)x0[1];
  a[base + 2] = (_Float16)x0[2]; a[base + 3] = (_Float16)x0[3];
  a[base + 4] = (_Float16)x1[0]; a[base + 5] = (_Float16)x1[1];
  a[base + 6] = (_Float16)x1[2]; a[base + 7] = (_Float16)x1[3];
}

// ---------------------------------------------------------------------------
// Kernel 1: weights[n] = silu(feat[n,:] @ W1 + b1) @ W2 + b2  (WMMA f16)
//           denom[idx[n]] += weights[n]  (atomic)
// One wave per 16-row tile; 8 waves / block; grid-stride over tiles.
// The N-tile loop is kept as a REAL loop (#pragma unroll 1) so only one C
// accumulator + 4 transient B fragments are live at a time -> no spills under
// the 8-waves/SIMD VGPR budget; B fragments stream from LDS per use.
// ---------------------------------------------------------------------------
__launch_bounds__(256, 8)
__global__ void mlp_weights_kernel(const float* __restrict__ feat,
                                   const float* __restrict__ W1,
                                   const float* __restrict__ b1,
                                   const float* __restrict__ W2,
                                   const float* __restrict__ b2,
                                   const int*   __restrict__ idx,
                                   float* __restrict__ weights,
                                   float* __restrict__ denom,
                                   int n, int ntiles) {
  // W1 staged in B-fragment order: [(ntile*4+kchunk)*32 + lane][16 halves]
  __shared__ __align__(32) _Float16 w1frag[8 * 4 * 32][16];
  __shared__ float s_b1[DIM];
  __shared__ float s_w2[DIM];

  const int lane = threadIdx.x & 31;
  const int wave = threadIdx.x >> 5;

  // B-fragment layout (16-bit B, 32x16): lane holds column N = lane&15;
  // lanes 0-15 hold K = kchunk*32 + 0..15, lanes 16-31 hold K = +16..31,
  // 16 consecutive K values per lane (two ds_load_b128 at use time).
  for (int e = threadIdx.x; e < 8 * 4 * 32; e += blockDim.x) {
    int t = e >> 7;          // N-tile (0..7)
    int c = (e >> 5) & 3;    // K-chunk (0..3)
    int l = e & 31;          // lane
    int col   = t * 16 + (l & 15);
    int kbase = c * 32 + (l >> 4) * 16;
#pragma unroll
    for (int i = 0; i < 16; ++i)
      w1frag[e][i] = (_Float16)W1[(kbase + i) * DIM + col];
  }
  if (threadIdx.x < DIM) {
    s_b1[threadIdx.x] = b1[threadIdx.x];
    s_w2[threadIdx.x] = W2[threadIdx.x];   // W2 is [128,1] -> flat 128
  }
  __syncthreads();

  const float b2v     = b2[0];
  const int   tstride = gridDim.x * 8;
  const _Float16* bbase = &w1frag[lane][0];   // per-lane base; +t*4096B/+c*1024B

  for (int tile = blockIdx.x * 8 + wave; tile < ntiles; tile += tstride) {
    const long rowbase = (long)tile * 16;
    int row = (int)rowbase + (lane & 15);
    if (row > n - 1) row = n - 1;            // clamp: EXEC must stay all-1s
    const float* fr = feat + (long)row * DIM;

    // Prefetch this lane's row slice of the next grid-stride tile.
    {
      long nrow = rowbase + (long)tstride * 16 + (lane & 15);
      if (nrow < n) __builtin_prefetch(feat + nrow * DIM, 0, 1);
    }

    // A-fragment layout (16-bit A, 16x32): lane holds row M = lane&15;
    // halves 0..7  = K = kc*32 + (lane>>4)*16 + 0..7
    // halves 8..15 = K = kc*32 + (lane>>4)*16 + 8..15
    v16h A[4];
#pragma unroll
    for (int c = 0; c < 4; ++c) {
      int k0 = c * 32 + (lane >> 4) * 16;
      load8_cvt_nt(A[c], 0, fr + k0);
      load8_cvt_nt(A[c], 8, fr + k0 + 8);
    }

    // Per-lane partial row-weights: wacc[r] belongs to row M = r + 8*(lane>>4)
    float wacc[8];
#pragma unroll
    for (int r = 0; r < 8; ++r) wacc[r] = 0.0f;

#pragma unroll 1                             // real loop: one C tile live
    for (int t = 0; t < 8; ++t) {
      v8f C = {};
#pragma unroll
      for (int c = 0; c < 4; ++c) {
        v16h Bf = *reinterpret_cast<const v16h*>(bbase + (t * 4 + c) * 32 * 16);
        C = __builtin_amdgcn_wmma_f32_16x16x32_f16(
            /*neg_a=*/false, A[c], /*neg_b=*/false, Bf,
            /*c_mod=*/(short)0, C, /*reuse_a=*/false, /*reuse_b=*/false);
      }
      // C layout: vgpr r, lanes 0-15 -> (M=r, N=lane); lanes 16-31 -> (M=r+8)
      const int   col = t * 16 + (lane & 15);
      const float b1c = s_b1[col];
      const float w2c = s_w2[col];
#pragma unroll
      for (int r = 0; r < 8; ++r) {
        float x = C[r] + b1c;
        // fast SiLU: x * rcp(1 + exp(-x))  (v_exp_f32 + v_rcp_f32)
        float s = x * __builtin_amdgcn_rcpf(1.0f + __expf(-x));
        wacc[r] += s * w2c;
      }
    }

    // Reduce across the 16 lanes that share each row (halves stay disjoint).
#pragma unroll
    for (int r = 0; r < 8; ++r) {
      float v = wacc[r];
      v += __shfl_xor(v, 1, 32);
      v += __shfl_xor(v, 2, 32);
      v += __shfl_xor(v, 4, 32);
      v += __shfl_xor(v, 8, 32);
      wacc[r] = v;
    }

    if ((lane & 15) == 0) {                  // lane 0 -> rows 0..7, lane 16 -> 8..15
      const long rb = rowbase + (long)(lane >> 4) * 8;
#pragma unroll
      for (int r = 0; r < 8; ++r) {
        long rr = rb + r;
        if (rr < n) {
          float w = wacc[r] + b2v;
          weights[rr] = w;
          atomicAdd(&denom[idx[rr]], w);
        }
      }
    }
  }
}

// ---------------------------------------------------------------------------
// Kernel 2: out[idx[n], :] += feat[n, :] * (weights[n] / denom[idx[n]])
// One wave per row; each lane handles 4 columns (non-temporal float4 load +
// 4 f32 atomics; the 32MB output accumulator stays hot in the 192MB L2).
// ---------------------------------------------------------------------------
__launch_bounds__(256)
__global__ void scatter_kernel(const float* __restrict__ feat,
                               const int*   __restrict__ idx,
                               const float* __restrict__ weights,
                               const float* __restrict__ denom,
                               float* __restrict__ out, int n) {
  const int lane = threadIdx.x & 31;
  const int wave = threadIdx.x >> 5;
  long row = (long)blockIdx.x * 8 + wave;
  if (row >= n) return;

  int   seg = idx[row];
  float nw  = weights[row] / denom[seg];   // exact IEEE divide, once per row

  const vf4 v = __builtin_nontemporal_load(
      reinterpret_cast<const vf4*>(feat + row * DIM) + lane);
  float* o = out + (long)seg * DIM + lane * 4;
  atomicAdd(o + 0, v[0] * nw);
  atomicAdd(o + 1, v[1] * nw);
  atomicAdd(o + 2, v[2] * nw);
  atomicAdd(o + 3, v[3] * nw);
}

// ---------------------------------------------------------------------------
// Launch: inputs = {features, W1, b1, W2, b2, ensemble_index, num_segments}
// ---------------------------------------------------------------------------
extern "C" void kernel_launch(void* const* d_in, const int* in_sizes, int n_in,
                              void* d_out, int out_size, void* d_ws, size_t ws_size,
                              hipStream_t stream) {
  const float* feat = (const float*)d_in[0];
  const float* W1   = (const float*)d_in[1];
  const float* b1   = (const float*)d_in[2];
  const float* W2   = (const float*)d_in[3];
  const float* b2   = (const float*)d_in[4];
  const int*   idx  = (const int*)d_in[5];

  const int N = in_sizes[0] / DIM;      // 1,000,000
  const int E = out_size   / DIM;       // 62,500
  float* out     = (float*)d_out;
  float* weights = (float*)d_ws;        // N floats
  float* denom   = weights + N;         // E floats

  // 1) zero output + denom
  {
    long nz = (long)E * DIM;
    long g  = (nz + 255) / 256;
    zero_kernel<<<(unsigned)g, 256, 0, stream>>>(out, nz, denom, E);
  }

  // 2) MLP weights via WMMA + denom atomics
  {
    int ntiles = (N + 15) / 16;
    int blocks = (ntiles + 7) / 8;
    if (blocks > 2048) blocks = 2048;   // grid-stride; amortize W1 LDS staging
    mlp_weights_kernel<<<blocks, 256, 0, stream>>>(feat, W1, b1, W2, b2, idx,
                                                   weights, denom, N, ntiles);
  }

  // 3) normalize + weighted scatter
  {
    int blocks = (N + 7) / 8;
    scatter_kernel<<<blocks, 256, 0, stream>>>(feat, idx, weights, denom, out, N);
  }
}
